// Transformer_block_68959994904799
// MI455X (gfx1250) — matrix operand order repaired and checked by reading, not compiled, not run on hardware
//
#include <hip/hip_runtime.h>
#include <stdint.h>

#define BB   2
#define TT   2048
#define EE   768
#define NH   12
#define HD   64
#define FFN  3072
#define MR   (BB*TT)      // 4096 rows
#define QT   (TT/16)      // 128 query tiles per (b,h)

typedef __bf16        v16bf __attribute__((ext_vector_type(16)));
typedef float         v8f   __attribute__((ext_vector_type(8)));
typedef unsigned int  u32x4 __attribute__((ext_vector_type(4)));
typedef unsigned int  u32x8 __attribute__((ext_vector_type(8)));

struct U256 { u32x4 lo, hi; };

__device__ __forceinline__ unsigned short f2bf(float f) {
    unsigned int u = __float_as_uint(f);
    unsigned int r = (u + 0x7FFFu + ((u >> 16) & 1u)) >> 16;   // RNE
    return (unsigned short)r;
}

// Build a v16bf fragment from two 16-byte chunks (global or LDS).
__device__ __forceinline__ v16bf load_frag(const unsigned short* p0,
                                           const unsigned short* p1) {
    U256 u;
    u.lo = *(const u32x4*)p0;
    u.hi = *(const u32x4*)p1;
    return __builtin_bit_cast(v16bf, u);
}

__device__ __forceinline__ v8f wmma_bf16(v16bf a, v16bf b, v8f c) {
    return __builtin_amdgcn_wmma_f32_16x16x32_bf16(
        false, a, false, b, (short)0, c, false, false);
}

// ---------------------------------------------------------------- TDM
// Issue a 2D tensor_load_to_lds: tile 32 (K halves) x 128 (rows) of bf16,
// source row stride K elements, dest rows padded 64B+16B (bank-conflict-free).
// D# layout per CDNA5 ISA.
__device__ __forceinline__ void tdm_issue_b(unsigned ldsOff,
                                            const unsigned short* gsrc,
                                            int K, int N) {
    unsigned long long ga = (unsigned long long)(uintptr_t)gsrc;
    u32x4 g0;
    g0[0] = 1u;                                   // count=1, user descriptor
    g0[1] = ldsOff;                               // lds_addr (bytes)
    g0[2] = (unsigned)ga;                         // global_addr[31:0]
    g0[3] = ((unsigned)(ga >> 32) & 0x01FFFFFFu)  // global_addr[56:32]
          | (2u << 30);                           // type = 2 (image)
    u32x8 g1;
    g1[0] = (1u << 16)      // data_size = 1 -> 2 bytes
          | (1u << 20)      // pad_enable
          | (3u << 22)      // pad_interval = 3 -> every 16 DWORDs (one row)
          | (3u << 25);     // pad_amount   = 3 -> 4 DWORDs (16B)
    g1[1] = ((unsigned)K & 0xFFFFu) << 16;                    // tensor_dim0 lo
    g1[2] = ((unsigned)K >> 16) | (((unsigned)N & 0xFFFFu) << 16); // dim0 hi | dim1 lo
    g1[3] = ((unsigned)N >> 16) | (32u << 16);                // dim1 hi | tile_dim0=32
    g1[4] = 128u;                                             // tile_dim1=128, tile_dim2=0
    g1[5] = (unsigned)K;                                      // tensor_dim0_stride lo
    g1[6] = 0u;                                               // stride hi | dim1_stride lo
    g1[7] = 0u;
    asm volatile("tensor_load_to_lds %0, %1" :: "s"(g0), "s"(g1) : "memory");
}

// ---------------------------------------------------------------- convert
__global__ __launch_bounds__(256) void cvt_f32_bf16(const float* __restrict__ in,
                                                    unsigned short* __restrict__ out,
                                                    int n) {
    int i = blockIdx.x * 256 + threadIdx.x;
    if (i < n) out[i] = f2bf(in[i]);
}

// ---------------------------------------------------------------- GEMM
// y = A(bf16,[M,K]) * Bw(bf16,[N,K])^T ; wave tile 16x64, block 64x128.
// B tiles staged into LDS by the TDM, double-buffered, 80B padded rows.
// MODE 0: +bias +resid -> f32 out        (proj / ffn2)
// MODE 1: +bias, ReLU   -> bf16 out      (ffn1)
// MODE 3: scatter bf16 to [b,h,t,d]      (q,k)
// MODE 4: scatter bf16 to [b,h,d,t]      (v transposed)
#define BROW 40                 // halves per padded LDS row (64B data + 16B pad)
#define BUFH (128 * BROW)       // halves per buffer (10,240 B)

template<int MODE>
__global__ __launch_bounds__(256) void gemm_bf16_wmma(
    const unsigned short* __restrict__ A, const unsigned short* __restrict__ Bw,
    const float* __restrict__ bias, const float* __restrict__ resid,
    float* __restrict__ outf, unsigned short* __restrict__ outh,
    int M, int N, int K)
{
    __shared__ __align__(16) unsigned short btile[2][BUFH];
    const int lane = threadIdx.x & 31;
    const int wid  = threadIdx.x >> 5;
    const int wn   = wid & 1;
    const int m0   = blockIdx.y * 64 + (wid >> 1) * 16;
    const int n0b  = blockIdx.x * 128;
    const int l15  = lane & 15, lhi = lane >> 4;

    v8f c[4] = {{}, {}, {}, {}};
    const unsigned short* arow = A  + (size_t)(m0 + l15) * K + lhi * 8;
    const unsigned short* bblk = Bw + (size_t)n0b * K;
    const unsigned ldsOff0 = (unsigned)(uintptr_t)&btile[0][0];
    const unsigned ldsOff1 = (unsigned)(uintptr_t)&btile[1][0];

    const int nsteps = K >> 5;
    if (wid == 0) {
        tdm_issue_b(ldsOff0, bblk,      K, N);
        tdm_issue_b(ldsOff1, bblk + 32, K, N);
    }
    for (int step = 0; step < nsteps; ++step) {
        const int cur = step & 1;
        if (wid == 0) {
            if (step + 1 < nsteps) __builtin_amdgcn_s_wait_tensorcnt(1);
            else                   __builtin_amdgcn_s_wait_tensorcnt(0);
        }
        __syncthreads();                 // current buffer visible to all waves
        const int k0 = step << 5;
        v16bf a = load_frag(arow + k0, arow + k0 + 16);
#pragma unroll
        for (int t = 0; t < 4; t++) {
            const unsigned short* bp =
                &btile[cur][(wn * 64 + t * 16 + l15) * BROW + lhi * 8];
            v16bf b = load_frag(bp, bp + 16);
            c[t] = wmma_bf16(a, b, c[t]);
        }
        __syncthreads();                 // all waves consumed btile[cur]
        if (wid == 0 && step + 2 < nsteps)
            tdm_issue_b(cur ? ldsOff1 : ldsOff0,
                        bblk + (size_t)(step + 2) * 32, K, N);
    }

#pragma unroll
    for (int t = 0; t < 4; t++) {
#pragma unroll
        for (int r = 0; r < 8; r++) {
            int m = m0 + r + (lhi << 3);
            int n = n0b + wn * 64 + t * 16 + l15;
            float v = c[t][r];
            if (MODE == 0) {
                v += bias[n] + resid[(size_t)m * N + n];
                outf[(size_t)m * N + n] = v;
            } else if (MODE == 1) {
                v += bias[n];
                v = v > 0.0f ? v : 0.0f;
                outh[(size_t)m * N + n] = f2bf(v);
            } else if (MODE == 3) {
                int b = m >> 11, tq = m & 2047, h = n >> 6, d = n & 63;
                outh[(((size_t)(b * NH + h)) * TT + tq) * HD + d] = f2bf(v);
            } else if (MODE == 4) {
                int b = m >> 11, tq = m & 2047, h = n >> 6, d = n & 63;
                outh[(((size_t)(b * NH + h)) * HD + d) * TT + tq] = f2bf(v);
            }
        }
    }
}

// ---------------------------------------------------------------- attention
// One wave = one (b, h, 16-query tile). Flash-style online softmax.
__global__ __launch_bounds__(256) void attn_flash(
    const unsigned short* __restrict__ qb,   // [b,h,t,d] bf16
    const unsigned short* __restrict__ kb,   // [b,h,t,d] bf16
    const unsigned short* __restrict__ vtb,  // [b,h,d,t] bf16
    unsigned short* __restrict__ attnb)      // [b*t, h*d] bf16
{
    __shared__ __align__(16) unsigned short plds[8 * 16 * 32]; // 1KB per wave
    const int lane  = threadIdx.x & 31;
    const int wslot = threadIdx.x >> 5;
    const int wid   = blockIdx.x * 8 + wslot;
    const int b   = wid / (NH * QT);
    const int rem = wid % (NH * QT);
    const int h   = rem / QT;
    const int m0  = (rem % QT) * 16;
    const int l15 = lane & 15, lhi = lane >> 4;

    const unsigned short* qbase =
        qb + (((size_t)(b * NH + h)) * TT + m0 + l15) * HD + lhi * 8;
    v16bf aq0 = load_frag(qbase,      qbase + 16);   // K = 0..31
    v16bf aq1 = load_frag(qbase + 32, qbase + 48);   // K = 32..63

    const unsigned short* kbase = kb  + ((size_t)(b * NH + h)) * TT * HD;
    const unsigned short* vbase = vtb + ((size_t)(b * NH + h)) * HD * TT;
    unsigned short* myp = plds + wslot * 512;

    float rmax[8], rsum[8];
    v8f O[4] = {{}, {}, {}, {}};
#pragma unroll
    for (int r = 0; r < 8; r++) { rmax[r] = -3e38f; rsum[r] = 0.0f; }

    const int nEnd = m0 + 16;                 // causal: keys < m0+16
    for (int nb0 = 0; nb0 < nEnd; nb0 += 32) {
        float p[2][8], tmax[8];
#pragma unroll
        for (int r = 0; r < 8; r++) tmax[r] = -3e38f;

#pragma unroll
        for (int j = 0; j < 2; j++) {
            const unsigned short* krow =
                kbase + (size_t)(nb0 + j * 16 + l15) * HD + lhi * 8;
            v16bf b0 = load_frag(krow,      krow + 16);   // d = 0..31
            v16bf b1 = load_frag(krow + 32, krow + 48);   // d = 32..63
            v8f s = {};
            s = wmma_bf16(aq0, b0, s);
            s = wmma_bf16(aq1, b1, s);
#pragma unroll
            for (int r = 0; r < 8; r++) {
                float sv = s[r] * 0.125f;                 // 1/sqrt(64)
                int mq = m0 + r + (lhi << 3);
                int nk = nb0 + j * 16 + l15;
                if (nk > mq) sv = -1e30f;                 // causal mask
                p[j][r] = sv;
                tmax[r] = fmaxf(tmax[r], sv);
            }
        }
        // row max across the 16 lanes of each half-wave
#pragma unroll
        for (int r = 0; r < 8; r++) {
            tmax[r] = fmaxf(tmax[r], __shfl_xor(tmax[r], 1));
            tmax[r] = fmaxf(tmax[r], __shfl_xor(tmax[r], 2));
            tmax[r] = fmaxf(tmax[r], __shfl_xor(tmax[r], 4));
            tmax[r] = fmaxf(tmax[r], __shfl_xor(tmax[r], 8));
        }
        float alpha[8], psum[8];
#pragma unroll
        for (int r = 0; r < 8; r++) {
            float nm = fmaxf(rmax[r], tmax[r]);
            alpha[r] = __expf(rmax[r] - nm);
            rmax[r]  = nm;
            psum[r]  = 0.0f;
        }
#pragma unroll
        for (int j = 0; j < 2; j++)
#pragma unroll
            for (int r = 0; r < 8; r++) {
                p[j][r] = __expf(p[j][r] - rmax[r]);
                psum[r] += p[j][r];
            }
#pragma unroll
        for (int r = 0; r < 8; r++) {
            psum[r] += __shfl_xor(psum[r], 1);
            psum[r] += __shfl_xor(psum[r], 2);
            psum[r] += __shfl_xor(psum[r], 4);
            psum[r] += __shfl_xor(psum[r], 8);
            rsum[r] = rsum[r] * alpha[r] + psum[r];
        }
#pragma unroll
        for (int t = 0; t < 4; t++)
#pragma unroll
            for (int r = 0; r < 8; r++) O[t][r] *= alpha[r];

        // transpose P (C layout) -> A layout via per-wave LDS slab
#pragma unroll
        for (int j = 0; j < 2; j++)
#pragma unroll
            for (int r = 0; r < 8; r++)
                myp[(r + (lhi << 3)) * 32 + j * 16 + l15] = f2bf(p[j][r]);
        asm volatile("s_wait_dscnt 0" ::: "memory");
        const unsigned short* pr = myp + l15 * 32 + lhi * 8;
        v16bf pa = load_frag(pr, pr + 16);

        // O += P * V  (V^T stored, so B frags are contiguous)
#pragma unroll
        for (int t = 0; t < 4; t++) {
            const unsigned short* vrow =
                vbase + (size_t)(t * 16 + l15) * TT + nb0 + lhi * 8;
            v16bf vb = load_frag(vrow, vrow + 16);
            O[t] = wmma_bf16(pa, vb, O[t]);
        }
    }

#pragma unroll
    for (int r = 0; r < 8; r++) rsum[r] = 1.0f / rsum[r];
#pragma unroll
    for (int t = 0; t < 4; t++)
#pragma unroll
        for (int r = 0; r < 8; r++) {
            int mq  = m0 + r + (lhi << 3);
            int col = h * HD + t * 16 + l15;
            attnb[(size_t)(b * TT + mq) * EE + col] = f2bf(O[t][r] * rsum[r]);
        }
}

// ---------------------------------------------------------------- layernorm
__device__ __forceinline__ float block_sum(float v, float* red) {
    v += __shfl_xor(v, 1);  v += __shfl_xor(v, 2);
    v += __shfl_xor(v, 4);  v += __shfl_xor(v, 8);
    v += __shfl_xor(v, 16);
    if ((threadIdx.x & 31) == 0) red[threadIdx.x >> 5] = v;
    __syncthreads();
    float t = 0.0f;
#pragma unroll
    for (int i = 0; i < 8; i++) t += red[i];
    __syncthreads();
    return t;
}

__global__ __launch_bounds__(256) void layernorm_k(
    const float* __restrict__ in, const float* __restrict__ g,
    const float* __restrict__ bt, float* __restrict__ outf,
    unsigned short* __restrict__ outh)
{
    __shared__ float red[8];
    const size_t row = blockIdx.x;
    const float* x = in + row * EE;
    float v0 = x[threadIdx.x], v1 = x[threadIdx.x + 256], v2 = x[threadIdx.x + 512];
    float mu = block_sum(v0 + v1 + v2, red) * (1.0f / EE);
    float d0 = v0 - mu, d1 = v1 - mu, d2 = v2 - mu;
    float var = block_sum(d0 * d0 + d1 * d1 + d2 * d2, red) * (1.0f / EE);
    float rs = rsqrtf(var + 1e-5f);
    float dv[3] = {d0, d1, d2};
#pragma unroll
    for (int i = 0; i < 3; i++) {
        int c = threadIdx.x + i * 256;
        float y = dv[i] * rs * g[c] + bt[c];
        outf[row * EE + c] = y;
        if (outh) outh[row * EE + c] = f2bf(y);
    }
}

// ---------------------------------------------------------------- launcher
extern "C" void kernel_launch(void* const* d_in, const int* in_sizes, int n_in,
                              void* d_out, int out_size, void* d_ws, size_t ws_size,
                              hipStream_t stream)
{
    const float* x      = (const float*)d_in[0];
    const float* Wq     = (const float*)d_in[1];
    const float* Wk     = (const float*)d_in[2];
    const float* Wv     = (const float*)d_in[3];
    const float* proj_w = (const float*)d_in[4];
    const float* proj_b = (const float*)d_in[5];
    const float* ln1_g  = (const float*)d_in[6];
    const float* ln1_b  = (const float*)d_in[7];
    const float* w1     = (const float*)d_in[8];
    const float* b1     = (const float*)d_in[9];
    const float* w2     = (const float*)d_in[10];
    const float* b2     = (const float*)d_in[11];
    const float* ln2_g  = (const float*)d_in[12];
    const float* ln2_b  = (const float*)d_in[13];
    float* out = (float*)d_out;

    char* ws = (char*)d_ws;
    size_t off = 0;
    auto alloc = [&](size_t bytes) -> void* {
        void* p = ws + off;
        off = (off + bytes + 255) & ~(size_t)255;
        return p;
    };
    unsigned short* xb   = (unsigned short*)alloc((size_t)MR * EE * 2);
    unsigned short* wqb  = (unsigned short*)alloc((size_t)EE * EE * 2);
    unsigned short* wkb  = (unsigned short*)alloc((size_t)EE * EE * 2);
    unsigned short* wvb  = (unsigned short*)alloc((size_t)EE * EE * 2);
    unsigned short* pwb  = (unsigned short*)alloc((size_t)EE * EE * 2);
    unsigned short* w1b  = (unsigned short*)alloc((size_t)FFN * EE * 2);
    unsigned short* w2b  = (unsigned short*)alloc((size_t)EE * FFN * 2);
    unsigned short* qbuf = (unsigned short*)alloc((size_t)MR * EE * 2);
    unsigned short* kbuf = (unsigned short*)alloc((size_t)MR * EE * 2);
    unsigned short* vtbf = (unsigned short*)alloc((size_t)MR * EE * 2);
    unsigned short* attb = (unsigned short*)alloc((size_t)MR * EE * 2);
    unsigned short* x1b  = (unsigned short*)alloc((size_t)MR * EE * 2);
    unsigned short* hb   = (unsigned short*)alloc((size_t)MR * FFN * 2);
    float* y1 = (float*)alloc((size_t)MR * EE * 4);
    float* x1 = (float*)alloc((size_t)MR * EE * 4);
    float* y2 = (float*)alloc((size_t)MR * EE * 4);

    auto cvt = [&](const float* src, unsigned short* dst, int n) {
        cvt_f32_bf16<<<(n + 255) / 256, 256, 0, stream>>>(src, dst, n);
    };
    cvt(x,      xb,  MR * EE);
    cvt(Wq,     wqb, EE * EE);
    cvt(Wk,     wkb, EE * EE);
    cvt(Wv,     wvb, EE * EE);
    cvt(proj_w, pwb, EE * EE);
    cvt(w1,     w1b, FFN * EE);
    cvt(w2,     w2b, EE * FFN);

    dim3 blk(256);
    dim3 gE(EE / 128, MR / 64);    // N=768
    dim3 gF(FFN / 128, MR / 64);   // N=3072

    // QKV projections (scatter to per-head layouts; V transposed)
    gemm_bf16_wmma<3><<<gE, blk, 0, stream>>>(xb, wqb, nullptr, nullptr, nullptr, qbuf, MR, EE, EE);
    gemm_bf16_wmma<3><<<gE, blk, 0, stream>>>(xb, wkb, nullptr, nullptr, nullptr, kbuf, MR, EE, EE);
    gemm_bf16_wmma<4><<<gE, blk, 0, stream>>>(xb, wvb, nullptr, nullptr, nullptr, vtbf, MR, EE, EE);

    // Flash attention: B*H*(T/16) = 3072 waves -> 384 blocks of 8 waves
    attn_flash<<<BB * NH * QT / 8, blk, 0, stream>>>(qbuf, kbuf, vtbf, attb);

    // out projection + bias + residual, then LN1
    gemm_bf16_wmma<0><<<gE, blk, 0, stream>>>(attb, pwb, proj_b, x, y1, nullptr, MR, EE, EE);
    layernorm_k<<<MR, blk, 0, stream>>>(y1, ln1_g, ln1_b, x1, x1b);

    // FFN
    gemm_bf16_wmma<1><<<gF, blk, 0, stream>>>(x1b, w1b, b1, nullptr, nullptr, hb, MR, FFN, EE);
    gemm_bf16_wmma<0><<<gE, blk, 0, stream>>>(hb, w2b, b2, x1, y2, nullptr, MR, EE, FFN);
    layernorm_k<<<MR, blk, 0, stream>>>(y2, ln2_g, ln2_b, out, nullptr);
}
